// CoarseMatching_73538430042970
// MI455X (gfx1250) — matrix operand order repaired and checked by reading, not compiled, not run on hardware
//
#include <hip/hip_runtime.h>
#include <hip/hip_bf16.h>

#define LDIM 4800
#define SDIM 4800
#define CDIM 256
#define SIM_SCALE 0.625f   // 1/(sqrt(256)*0.1)

typedef __bf16          v16bf __attribute__((ext_vector_type(16)));
typedef float           v8f   __attribute__((ext_vector_type(8)));
typedef unsigned short  u16x8 __attribute__((ext_vector_type(8)));

union Frag {
  v16bf v;
  u16x8 h[2];
};

__device__ __forceinline__ unsigned short f2bf(float f) {
  // round-to-nearest-even fp32 -> bf16 (integer path; always compiles)
  unsigned int x = __float_as_uint(f);
  unsigned int r = x + 0x7FFFu + ((x >> 16) & 1u);
  return (unsigned short)(r >> 16);
}

// ---------------------------------------------------------------------------
// Kernel 0: one-time fp32 -> bf16 conversion of a feature matrix (off the
// GEMM critical path; 2.46 MB per matrix, negligible traffic).
// ---------------------------------------------------------------------------
__global__ __launch_bounds__(256) void k0_cvt_bf16(const float* __restrict__ in,
                                                   unsigned short* __restrict__ out,
                                                   int n) {
  const int i = blockIdx.x * 256 + threadIdx.x;
  if (i < n) out[i] = f2bf(in[i]);
}

// ---------------------------------------------------------------------------
// Kernel 1: sim = scale * A(LxC) * B(SxC)^T  via v_wmma_f32_16x16x32_bf16.
// One wave computes a 32x32 tile: 2 A-frags x 2 B-frags, 4 WMMA per K-step.
// Operands are pre-converted bf16; fragment loads are aligned b128 vectors.
// ---------------------------------------------------------------------------
__global__ __launch_bounds__(256) void k1_gemm_sim(const unsigned short* __restrict__ A,
                                                   const unsigned short* __restrict__ B,
                                                   float* __restrict__ sim) {
  const int wave = threadIdx.x >> 5;
  const int lane = threadIdx.x & 31;
  const int TN = LDIM / 32;                 // 150 tiles per dimension
  const int tile = blockIdx.x * 8 + wave;   // wave-uniform guard (EXEC all-1s)
  if (tile >= TN * TN) return;
  const int tm = tile / TN;
  const int tn = tile - tm * TN;
  const int m0 = tm * 32, n0 = tn * 32;
  const int m  = lane & 15;
  const int hi = lane >> 4;

  const unsigned short* a0p = A + (size_t)(m0 + m) * CDIM;
  const unsigned short* a1p = A + (size_t)(m0 + 16 + m) * CDIM;
  const unsigned short* b0p = B + (size_t)(n0 + m) * CDIM;
  const unsigned short* b1p = B + (size_t)(n0 + 16 + m) * CDIM;

  v8f c00 = {}, c01 = {}, c10 = {}, c11 = {};

  for (int k0 = 0; k0 < CDIM; k0 += 32) {
    Frag a0, a1, b0, b1;
    // 16-bit A 16x32 layout: lane(m,hi) holds K = k0+8*hi+t (u[0..7])
    // and K = k0+16+8*hi+t (u[8..15]); both runs 16B-aligned.
    const int ak = k0 + 8 * hi;
    a0.h[0] = *(const u16x8*)(a0p + ak);
    a0.h[1] = *(const u16x8*)(a0p + ak + 16);
    a1.h[0] = *(const u16x8*)(a1p + ak);
    a1.h[1] = *(const u16x8*)(a1p + ak + 16);
    // Dense B 32x16 layout: lane(m,hi) holds N = col m, K = k0+8*hi+t (u[0..7])
    // and K = k0+16+8*hi+t (u[8..15]). B element (k,n) = feat_c1[n][k].
    const int bk = k0 + 8 * hi;
    b0.h[0] = *(const u16x8*)(b0p + bk);
    b0.h[1] = *(const u16x8*)(b0p + bk + 16);
    b1.h[0] = *(const u16x8*)(b1p + bk);
    b1.h[1] = *(const u16x8*)(b1p + bk + 16);

    c00 = __builtin_amdgcn_wmma_f32_16x16x32_bf16(false, a0.v, false, b0.v,
                                                  (short)0, c00, false, false);
    c01 = __builtin_amdgcn_wmma_f32_16x16x32_bf16(false, a0.v, false, b1.v,
                                                  (short)0, c01, false, false);
    c10 = __builtin_amdgcn_wmma_f32_16x16x32_bf16(false, a1.v, false, b0.v,
                                                  (short)0, c10, false, false);
    c11 = __builtin_amdgcn_wmma_f32_16x16x32_bf16(false, a1.v, false, b1.v,
                                                  (short)0, c11, false, false);
  }

  // C/D layout: lane(m,hi), VGPR v -> (M = v + 8*hi, N = m)
#pragma unroll
  for (int v = 0; v < 8; ++v) {
    const int r = v + 8 * hi;
    sim[(size_t)(m0 + r)      * SDIM + (n0 + m)]      = c00[v] * SIM_SCALE;
    sim[(size_t)(m0 + r)      * SDIM + (n0 + 16 + m)] = c01[v] * SIM_SCALE;
    sim[(size_t)(m0 + 16 + r) * SDIM + (n0 + m)]      = c10[v] * SIM_SCALE;
    sim[(size_t)(m0 + 16 + r) * SDIM + (n0 + 16 + m)] = c11[v] * SIM_SCALE;
  }
}

__device__ __forceinline__ void online_update(float x, float& m, float& s) {
  if (x > m) { s = s * __expf(m - x) + 1.0f; m = x; }
  else       { s += __expf(x - m); }
}

// ---------------------------------------------------------------------------
// Kernel 2: per-row max & sum(exp(s - max))   (softmax over axis=2)
// ---------------------------------------------------------------------------
__global__ __launch_bounds__(256) void k2_row_stats(const float* __restrict__ sim,
                                                    float* __restrict__ rowmax,
                                                    float* __restrict__ rowsum) {
  const int i = blockIdx.x;
  const float* row = sim + (size_t)i * SDIM;
  float m = -1e30f, s = 0.0f;
  for (int j = threadIdx.x; j < SDIM; j += 256) online_update(row[j], m, s);

  __shared__ float sm[256], ss[256];
  sm[threadIdx.x] = m; ss[threadIdx.x] = s;
  __syncthreads();
  for (int off = 128; off > 0; off >>= 1) {
    if ((int)threadIdx.x < off) {
      float m1 = sm[threadIdx.x], s1 = ss[threadIdx.x];
      float m2 = sm[threadIdx.x + off], s2 = ss[threadIdx.x + off];
      float M = fmaxf(m1, m2);
      sm[threadIdx.x] = M;
      ss[threadIdx.x] = s1 * __expf(m1 - M) + s2 * __expf(m2 - M);
    }
    __syncthreads();
  }
  if (threadIdx.x == 0) { rowmax[i] = sm[0]; rowsum[i] = ss[0]; }
}

// ---------------------------------------------------------------------------
// Kernel 3: per-column max & sum(exp)   (softmax over axis=1)
// Block handles 64 columns; 4 row-quarters per column; coalesced reads.
// ---------------------------------------------------------------------------
__global__ __launch_bounds__(256) void k3_col_stats(const float* __restrict__ sim,
                                                    float* __restrict__ colmax,
                                                    float* __restrict__ colsum) {
  const int col = blockIdx.x * 64 + (threadIdx.x & 63);
  const int rq  = threadIdx.x >> 6;
  float m = -1e30f, s = 0.0f;
  for (int i = rq; i < LDIM; i += 4)
    online_update(sim[(size_t)i * SDIM + col], m, s);

  __shared__ float sm[256], ss[256];
  sm[threadIdx.x] = m; ss[threadIdx.x] = s;
  __syncthreads();
  if (rq == 0) {
    float M = sm[threadIdx.x], S = ss[threadIdx.x];
#pragma unroll
    for (int q = 1; q < 4; ++q) {
      float m2 = sm[threadIdx.x + 64 * q], s2 = ss[threadIdx.x + 64 * q];
      float Mn = fmaxf(M, m2);
      S = S * __expf(M - Mn) + s2 * __expf(m2 - Mn);
      M = Mn;
    }
    colmax[col] = M;
    colsum[col] = S;
  }
}

// ---------------------------------------------------------------------------
// Kernel 4: conf = softmax_col * softmax_row, write conf, fused row argmax.
// ---------------------------------------------------------------------------
__global__ __launch_bounds__(256) void k4_conf_argmax(const float* __restrict__ sim,
                                                      const float* __restrict__ rowmax,
                                                      const float* __restrict__ rowsum,
                                                      const float* __restrict__ colmax,
                                                      const float* __restrict__ colsum,
                                                      float* __restrict__ conf,
                                                      float* __restrict__ j_ids,
                                                      float* __restrict__ mconf) {
  const int i = blockIdx.x;
  const float rm = rowmax[i];
  const float rinv = 1.0f / rowsum[i];
  const float* srow = sim + (size_t)i * SDIM;
  float* crow = conf + (size_t)i * SDIM;

  float best = -1.0f; int bj = 0;
  for (int j = threadIdx.x; j < SDIM; j += 256) {
    float x = srow[j];
    float v = (__expf(x - rm) * rinv) * (__expf(x - colmax[j]) / colsum[j]);
    crow[j] = v;
    if (v > best) { best = v; bj = j; }
  }

  __shared__ float sv[256];
  __shared__ int   sj[256];
  sv[threadIdx.x] = best; sj[threadIdx.x] = bj;
  __syncthreads();
  for (int off = 128; off > 0; off >>= 1) {
    if ((int)threadIdx.x < off) {
      float v2 = sv[threadIdx.x + off]; int j2 = sj[threadIdx.x + off];
      if (v2 > sv[threadIdx.x] ||
          (v2 == sv[threadIdx.x] && j2 < sj[threadIdx.x])) {
        sv[threadIdx.x] = v2; sj[threadIdx.x] = j2;
      }
    }
    __syncthreads();
  }
  if (threadIdx.x == 0) {
    mconf[i] = sv[0];
    j_ids[i] = (float)sj[0];
  }
}

// ---------------------------------------------------------------------------
// Kernel 5: b_ids, i_ids, mkpts0_c, mkpts1_c from device-side scalars.
// ---------------------------------------------------------------------------
__global__ __launch_bounds__(256) void k5_tail(const int* __restrict__ hwi0,
                                               const int* __restrict__ h0c,
                                               const int* __restrict__ w0c,
                                               const int* __restrict__ w1c,
                                               const float* __restrict__ j_ids,
                                               float* __restrict__ b_ids,
                                               float* __restrict__ i_ids,
                                               float* __restrict__ mk0,
                                               float* __restrict__ mk1) {
  const int i = blockIdx.x * 256 + threadIdx.x;
  if (i >= LDIM) return;
  const float scale = (float)hwi0[0] / (float)h0c[0];
  const int W0 = w0c[0], W1 = w1c[0];
  b_ids[i] = 0.0f;
  i_ids[i] = (float)i;
  mk0[2 * i]     = (float)(i % W0) * scale;
  mk0[2 * i + 1] = (float)(i / W0) * scale;
  const int j = (int)j_ids[i];
  mk1[2 * i]     = (float)(j % W1) * scale;
  mk1[2 * i + 1] = (float)(j / W1) * scale;
}

extern "C" void kernel_launch(void* const* d_in, const int* in_sizes, int n_in,
                              void* d_out, int out_size, void* d_ws, size_t ws_size,
                              hipStream_t stream) {
  const float* f0  = (const float*)d_in[0];   // [1, 4800, 256]
  const float* f1  = (const float*)d_in[1];   // [1, 4800, 256]
  const int* hwi0  = (const int*)d_in[2];
  const int* h0c   = (const int*)d_in[3];
  const int* w0c   = (const int*)d_in[4];
  const int* w1c   = (const int*)d_in[5];

  float* out   = (float*)d_out;
  float* conf  = out;                                   // 4800*4800
  float* b_ids = conf + (size_t)LDIM * SDIM;            // 4800
  float* i_ids = b_ids + LDIM;                          // 4800
  float* j_ids = i_ids + LDIM;                          // 4800
  float* mk0   = j_ids + LDIM;                          // 9600
  float* mk1   = mk0 + 2 * LDIM;                        // 9600
  float* mconf = mk1 + 2 * LDIM;                        // 4800

  float* sim    = (float*)d_ws;                         // 4800*4800 f32
  float* rowmax = sim + (size_t)LDIM * SDIM;
  float* rowsum = rowmax + LDIM;
  float* colmax = rowsum + LDIM;
  float* colsum = colmax + SDIM;
  unsigned short* f0bf = (unsigned short*)(colsum + SDIM);  // 4800*256 bf16
  unsigned short* f1bf = f0bf + (size_t)LDIM * CDIM;        // 4800*256 bf16

  const int nfeat  = LDIM * CDIM;                       // 1,228,800
  const int ntiles = (LDIM / 32) * (SDIM / 32);         // 22500 wave-tiles
  const int nblk1  = (ntiles + 7) / 8;                  // 8 waves/block

  k0_cvt_bf16<<<(nfeat + 255) / 256, 256, 0, stream>>>(f0, f0bf, nfeat);
  k0_cvt_bf16<<<(nfeat + 255) / 256, 256, 0, stream>>>(f1, f1bf, nfeat);
  k1_gemm_sim<<<nblk1, 256, 0, stream>>>(f0bf, f1bf, sim);
  k2_row_stats<<<LDIM, 256, 0, stream>>>(sim, rowmax, rowsum);
  k3_col_stats<<<SDIM / 64, 256, 0, stream>>>(sim, colmax, colsum);
  k4_conf_argmax<<<LDIM, 256, 0, stream>>>(sim, rowmax, rowsum, colmax, colsum,
                                           conf, j_ids, mconf);
  k5_tail<<<(LDIM + 255) / 256, 256, 0, stream>>>(hwi0, h0c, w0c, w1c, j_ids,
                                                  b_ids, i_ids, mk0, mk1);
}
